// TFMambaBlock_27797028340370
// MI455X (gfx1250) — matrix operand order repaired and checked by reading, not compiled, not run on hardware
//
#include <hip/hip_runtime.h>
#include <math.h>

typedef __bf16 bf16;
typedef __attribute__((ext_vector_type(16))) __bf16 v16bf;
typedef __attribute__((ext_vector_type(8)))  __bf16 v8bf;
typedef __attribute__((ext_vector_type(8)))  float   v8f;

#define DEV static __device__ __forceinline__

DEV bf16 f2bf(float f) {
  unsigned u = __builtin_bit_cast(unsigned, f);
  unsigned r = (u + 0x7FFFu + ((u >> 16) & 1u)) >> 16;
  unsigned short h = (unsigned short)r;
  return __builtin_bit_cast(bf16, h);
}
DEV float sigmoidf_(float x) { return 1.0f / (1.0f + __expf(-x)); }
DEV float siluf_(float x)    { return x * sigmoidf_(x); }
DEV float softplusf_(float x){ return (x > 20.0f) ? x : log1pf(__expf(x)); }

// ---------------------------------------------------------------------------
// bf16 WMMA GEMM:  C(M,N) = A(M,K) * W(N,K)^T [+ bias]
// One wave per 16x16 tile, K-loop step 32, v_wmma_f32_16x16x32_bf16.
// M multiple of 16, K multiple of 32.
// FULLN=true  : N multiple of 16, no per-lane guards (no EXEC churn in loop)
// FULLN=false : ragged N, B loads and C stores guarded
// ---------------------------------------------------------------------------
template <bool FULLN>
__global__ void k_gemm_wmma(const bf16* __restrict__ A, const bf16* __restrict__ W,
                            const float* __restrict__ bias, float* __restrict__ C,
                            int M, int N, int K) {
  const int lane = threadIdx.x & 31;
  const int wid  = blockIdx.x * (blockDim.x >> 5) + (threadIdx.x >> 5);
  const int ntN  = (N + 15) >> 4;
  const int ntM  = M >> 4;
  if (wid >= ntM * ntN) return;                  // wave-uniform: EXEC stays full
  const int tm = wid / ntN, tn = wid % ntN;
  const int m16 = lane & 15;
  const int kh  = lane >> 4;                     // K-half select
  const int rowA = tm * 16 + m16;
  const int colB = tn * 16 + m16;
  const bool colOK = FULLN || (colB < N);

  const bf16* aptr = A + (size_t)rowA * K + 16 * kh;
  const bf16* bptr = W + (size_t)(colOK ? colB : 0) * K + 16 * kh;

  v8f acc = {};
  for (int kb = 0; kb < K; kb += 32) {
    // A fragment: lanes 0-15 hold K = kb+[0..15], lanes 16-31 K = kb+[16..31]
    v8bf alo = *reinterpret_cast<const v8bf*>(aptr + kb);
    v8bf ahi = *reinterpret_cast<const v8bf*>(aptr + kb + 8);
    v16bf afrag;
#pragma unroll
    for (int i = 0; i < 8; ++i) { afrag[i] = alo[i]; afrag[i + 8] = ahi[i]; }

    // B fragment: lane holds column colB, K = kb + 16*kh + [0..15] contiguous in W row
    v16bf bfrag = *reinterpret_cast<const v16bf*>(bptr + kb);
    if (!FULLN && !colOK) {
#pragma unroll
      for (int i = 0; i < 16; ++i) bfrag[i] = f2bf(0.0f);
    }
    if (kb + 32 < K) __builtin_prefetch(aptr + kb + 32, 0, 3);

    acc = __builtin_amdgcn_wmma_f32_16x16x32_bf16(
        false, afrag, false, bfrag, (short)0, acc, false, false);
  }

  const float bv = (bias && colOK) ? bias[colB] : 0.0f;
#pragma unroll
  for (int r = 0; r < 8; ++r) {
    int row = tm * 16 + kh * 8 + r;
    if (colOK) C[(size_t)row * N + colB] = acc[r] + bv;
  }
}

// ---------------------------------------------------------------------------
// Elementwise / transform kernels
// ---------------------------------------------------------------------------
__global__ void k_cvt_bf(const float* __restrict__ s, bf16* __restrict__ d, int n) {
  int i = blockIdx.x * blockDim.x + threadIdx.x;
  if (i < n) d[i] = f2bf(s[i]);
}
__global__ void k_negexp(const float* __restrict__ s, float* __restrict__ d, int n) {
  int i = blockIdx.x * blockDim.x + threadIdx.x;
  if (i < n) d[i] = -__expf(s[i]);
}
// fl_w (128,64) -> transposed bf16 (64,128): d[c*128+e] = fl_w[e*64+c]
__global__ void k_cvt_flwT(const float* __restrict__ s, bf16* __restrict__ d) {
  int i = blockIdx.x * blockDim.x + threadIdx.x;
  if (i < 64 * 128) { int c = i >> 7, e = i & 127; d[i] = f2bf(s[e * 64 + c]); }
}
// x (2,64,256,64) -> xt (2*64, 256, 64): xt[((b*64+f)*256+t)*64+c] = x[((b*64+c)*256+t)*64+f]
__global__ void k_t0(const float* __restrict__ x, float* __restrict__ xt) {
  int i = blockIdx.x * blockDim.x + threadIdx.x;
  if (i >= 2 * 64 * 256 * 64) return;
  int c = i & 63, r = i >> 6;
  int t = r & 255, q = r >> 8;
  int f = q & 63, b = q >> 6;
  xt[i] = x[(((size_t)(b * 64 + c) * 256 + t) * 64) + f];
}
// xt (2*64,256,64) -> xf (2*256,64,64)
__global__ void k_t1(const float* __restrict__ xt, float* __restrict__ xf) {
  int i = blockIdx.x * blockDim.x + threadIdx.x;
  if (i >= 2 * 64 * 256 * 64) return;
  int c = i & 63, r = i >> 6;
  int f = r & 63, q = r >> 6;
  int t = q & 255, b = q >> 8;
  xf[i] = xt[(((size_t)(b * 64 + f) * 256 + t) * 64) + c];
}
// flip along f: (512,64,64)
__global__ void k_flip(const float* __restrict__ s, float* __restrict__ d) {
  int i = blockIdx.x * blockDim.x + threadIdx.x;
  if (i >= 512 * 64 * 64) return;
  int c = i & 63, r = i >> 6;
  int f = r & 63, sq = r >> 6;
  d[(((size_t)sq * 64) + (63 - f)) * 64 + c] = s[i];
}
// RMSNorm over last dim (=64), writes bf16 * w
__global__ void k_rmsnorm(const float* __restrict__ x, const float* __restrict__ w,
                          bf16* __restrict__ o, int rows) {
  int row = blockIdx.x * blockDim.x + threadIdx.x;
  if (row >= rows) return;
  const float4* p4 = reinterpret_cast<const float4*>(x + (size_t)row * 64);
  float4 v[16]; float ss = 0.0f;
#pragma unroll
  for (int i = 0; i < 16; ++i) {
    v[i] = p4[i];
    ss += v[i].x * v[i].x + v[i].y * v[i].y + v[i].z * v[i].z + v[i].w * v[i].w;
  }
  float r = rsqrtf(ss * (1.0f / 64.0f) + 1e-5f);
  bf16* op = o + (size_t)row * 64;
#pragma unroll
  for (int i = 0; i < 16; ++i) {
    op[4 * i + 0] = f2bf(v[i].x * r * w[4 * i + 0]);
    op[4 * i + 1] = f2bf(v[i].y * r * w[4 * i + 1]);
    op[4 * i + 2] = f2bf(v[i].z * r * w[4 * i + 2]);
    op[4 * i + 3] = f2bf(v[i].w * r * w[4 * i + 3]);
  }
}
// causal depthwise conv (DC=4) + silu; xi is xz[:, :128] (stride 256)
__global__ void k_conv_silu(const float* __restrict__ xz, const float* __restrict__ cw,
                            const float* __restrict__ cb, float* __restrict__ xc,
                            bf16* __restrict__ xcb, int total, int L) {
  int i = blockIdx.x * blockDim.x + threadIdx.x;
  if (i >= total) return;
  int d = i & 127, m = i >> 7, l = m & (L - 1);   // L is a power of two (256 / 64)
  float s = cb[d];
#pragma unroll
  for (int k = 0; k < 4; ++k) {
    int ll = l - 3 + k;
    if (ll >= 0) s += cw[d * 4 + k] * xz[(size_t)(m - 3 + k) * 256 + d];
  }
  float v = siluf_(s);
  xc[i] = v; xcb[i] = f2bf(v);
}
// ---------------------------------------------------------------------------
// Fused selective scan: softplus(dt@dt_proj^T + bias) -> scan -> +u*D -> *silu(z)
// One block per sequence, thread = channel d (128 thr), 16-wide state in VGPRs.
// x_dbl row (36 floats: dt[4] | B[16] | C[16]) staged in LDS once per step.
// Writes the bf16 GEMM input for out_proj directly.
// ---------------------------------------------------------------------------
__global__ void k_scan_fused(const float* __restrict__ xdbl, const float* __restrict__ u,
                             const float* __restrict__ xz,   // z at col 128, stride 256
                             const float* __restrict__ Aneg, const float* __restrict__ Dw,
                             const float* __restrict__ dtw,  // dt_proj (128,4)
                             const float* __restrict__ dtb,  // dt_bias (128)
                             bf16* __restrict__ ybf, int L) {
  const int b = blockIdx.x, d = threadIdx.x;
  __shared__ float sX[36];
  float Ar[16], h[16], wr[4];
#pragma unroll
  for (int n = 0; n < 16; ++n) { Ar[n] = Aneg[d * 16 + n]; h[n] = 0.0f; }
#pragma unroll
  for (int r = 0; r < 4; ++r) wr[r] = dtw[d * 4 + r];
  const float bias = dtb[d];
  const float Dd = Dw[d];
  const size_t base = (size_t)b * L;
  for (int l = 0; l < L; ++l) {
    if (threadIdx.x < 36) sX[threadIdx.x] = xdbl[(base + l) * 36 + threadIdx.x];
    __syncthreads();
    float s = bias + sX[0] * wr[0] + sX[1] * wr[1] + sX[2] * wr[2] + sX[3] * wr[3];
    float dt = softplusf_(s);
    float ut = u[(base + l) * 128 + d];
    float zt = xz[(base + l) * 256 + 128 + d];
    float du = dt * ut;
    float acc = 0.0f;
#pragma unroll
    for (int n = 0; n < 16; ++n) {
      h[n] = h[n] * __expf(dt * Ar[n]) + du * sX[4 + n];
      acc += h[n] * sX[20 + n];
    }
    ybf[(base + l) * 128 + d] = f2bf((acc + ut * Dd) * siluf_(zt));
    __syncthreads();
  }
}
__global__ void k_add_scaled(float* __restrict__ dst, const float* __restrict__ a,
                             const float* __restrict__ b, float sb, int n) {
  int i = blockIdx.x * blockDim.x + threadIdx.x;
  if (i < n) dst[i] = a[i] + sb * b[i];
}
__global__ void k_concat(const float* __restrict__ yf, const float* __restrict__ yb,
                         bf16* __restrict__ cat, int rows) {
  int i = blockIdx.x * blockDim.x + threadIdx.x;
  if (i >= rows * 128) return;
  int j = i & 127, m = i >> 7;
  float v = (j < 64) ? yf[(size_t)m * 64 + j] : yb[(size_t)m * 64 + (j - 64)];
  cat[i] = f2bf(v);
}
// out[b,c,t,f] = yfin[(s,f),c] + xf[(s,f),c], s = b*256+t
__global__ void k_scatter(const float* __restrict__ yfin, const float* __restrict__ xf,
                          float* __restrict__ out, int total) {
  int i = blockIdx.x * blockDim.x + threadIdx.x;
  if (i >= total) return;
  int c = i & 63, m = i >> 6;
  int f = m & 63, s = m >> 6;
  int t = s & 255, b = s >> 8;
  out[(((size_t)(b * 64 + c) * 256 + t) * 64) + f] = yfin[i] + xf[i];
}

// ---------------------------------------------------------------------------
// Host-side orchestration
// ---------------------------------------------------------------------------
static inline int cdiv(int a, int b) { return (a + b - 1) / b; }

struct MambaBufs {
  float *xz, *xc, *xdbl, *outb;
  bf16 *rn_bf, *xc_bf, *ybf;
};

static void run_mamba(hipStream_t st, const float* const* P, const bf16* w_in,
                      const bf16* w_xp, const bf16* w_op, const float* Aneg,
                      const float* x_io, const MambaBufs& B, int Bseq, int L) {
  const int M = Bseq * L;                 // 32768 in all cases
  const int TPB = 256;
  // 1. rmsnorm -> bf16
  k_rmsnorm<<<cdiv(M, TPB), TPB, 0, st>>>(x_io, P[0], B.rn_bf, M);
  // 2. xz = rn @ in_proj^T (N=256, K=64)
  {
    int waves = (M / 16) * (256 / 16);
    k_gemm_wmma<true><<<cdiv(waves, 4), 128, 0, st>>>(B.rn_bf, w_in, nullptr, B.xz, M, 256, 64);
  }
  // 3. depthwise conv + silu
  k_conv_silu<<<cdiv(M * 128, TPB), TPB, 0, st>>>(B.xz, P[2], P[3], B.xc, B.xc_bf, M * 128, L);
  // 4. x_dbl = xc @ x_proj^T (N=36 ragged, K=128)
  {
    int waves = (M / 16) * cdiv(36, 16);
    k_gemm_wmma<false><<<cdiv(waves, 4), 128, 0, st>>>(B.xc_bf, w_xp, nullptr, B.xdbl, M, 36, 128);
  }
  // 5. fused: delta(softplus) + selective scan + u*D + silu(z) gate -> bf16
  k_scan_fused<<<Bseq, 128, 0, st>>>(B.xdbl, B.xc, B.xz, Aneg, P[8], P[5], P[6], B.ybf, L);
  // 6. out = y @ out_proj^T (N=64, K=128)
  {
    int waves = (M / 16) * (64 / 16);
    k_gemm_wmma<true><<<cdiv(waves, 4), 128, 0, st>>>(B.ybf, w_op, nullptr, B.outb, M, 64, 128);
  }
}

extern "C" void kernel_launch(void* const* d_in, const int* in_sizes, int n_in,
                              void* d_out, int out_size, void* d_ws, size_t ws_size,
                              hipStream_t stream) {
  (void)in_sizes; (void)n_in; (void)out_size; (void)ws_size;
  const float* x    = (const float*)d_in[0];
  const float* P[3][10];
  for (int s = 0; s < 3; ++s)
    for (int j = 0; j < 10; ++j) P[s][j] = (const float*)d_in[1 + s * 10 + j];
  const float* fl_w = (const float*)d_in[31];
  const float* fl_b = (const float*)d_in[32];
  float* out = (float*)d_out;

  const int M = 32768, NE = M * 64;       // rows; elements of (M,64)
  char* base = (char*)d_ws;
  size_t off = 0;
  auto alloc = [&](size_t bytes) -> void* {
    void* p = base + off; off = (off + bytes + 255) & ~(size_t)255; return p;
  };
  float* xt   = (float*)alloc((size_t)NE * 4);
  float* xf   = (float*)alloc((size_t)NE * 4);
  float* xff  = (float*)alloc((size_t)NE * 4);
  float* yf   = (float*)alloc((size_t)NE * 4);
  float* yb   = (float*)alloc((size_t)NE * 4);
  MambaBufs B;
  B.outb  = (float*)alloc((size_t)NE * 4);
  B.xz    = (float*)alloc((size_t)M * 256 * 4);
  B.xc    = (float*)alloc((size_t)M * 128 * 4);
  B.xdbl  = (float*)alloc((size_t)M * 36 * 4);
  B.rn_bf = (bf16*)alloc((size_t)M * 64 * 2);
  B.xc_bf = (bf16*)alloc((size_t)M * 128 * 2);
  B.ybf   = (bf16*)alloc((size_t)M * 128 * 2);
  bf16* cat_bf = (bf16*)alloc((size_t)M * 128 * 2);
  bf16 *w_in[3], *w_xp[3], *w_op[3];
  float* Aneg[3];
  for (int s = 0; s < 3; ++s) {
    w_in[s] = (bf16*)alloc(256 * 64 * 2);
    w_xp[s] = (bf16*)alloc(36 * 128 * 2);
    w_op[s] = (bf16*)alloc(64 * 128 * 2);
    Aneg[s] = (float*)alloc(128 * 16 * 4);
  }
  bf16* w_flT = (bf16*)alloc(64 * 128 * 2);

  const int TPB = 256;
  // --- weight prep (deterministic, recomputed per call) ---
  for (int s = 0; s < 3; ++s) {
    k_cvt_bf<<<cdiv(256 * 64, TPB), TPB, 0, stream>>>(P[s][1], w_in[s], 256 * 64);
    k_cvt_bf<<<cdiv(36 * 128, TPB), TPB, 0, stream>>>(P[s][4], w_xp[s], 36 * 128);
    k_cvt_bf<<<cdiv(64 * 128, TPB), TPB, 0, stream>>>(P[s][9], w_op[s], 64 * 128);
    k_negexp<<<cdiv(128 * 16, TPB), TPB, 0, stream>>>(P[s][7], Aneg[s], 128 * 16);
  }
  k_cvt_flwT<<<cdiv(64 * 128, TPB), TPB, 0, stream>>>(fl_w, w_flT);

  // --- time branch: (b*f, t=256, c) ---
  k_t0<<<cdiv(NE, TPB), TPB, 0, stream>>>(x, xt);
  run_mamba(stream, P[0], w_in[0], w_xp[0], w_op[0], Aneg[0], xt, B, 128, 256);
  k_add_scaled<<<cdiv(NE, TPB), TPB, 0, stream>>>(xt, B.outb, xt, 2.0f, NE); // block + residual + outer skip

  // --- frequency branches: (b*t, f=64, c) ---
  k_t1<<<cdiv(NE, TPB), TPB, 0, stream>>>(xt, xf);
  run_mamba(stream, P[1], w_in[1], w_xp[1], w_op[1], Aneg[1], xf, B, 512, 64);
  k_add_scaled<<<cdiv(NE, TPB), TPB, 0, stream>>>(yf, B.outb, xf, 1.0f, NE);

  k_flip<<<cdiv(NE, TPB), TPB, 0, stream>>>(xf, xff);
  run_mamba(stream, P[2], w_in[2], w_xp[2], w_op[2], Aneg[2], xff, B, 512, 64);
  k_add_scaled<<<cdiv(NE, TPB), TPB, 0, stream>>>(B.outb, B.outb, xff, 1.0f, NE);
  k_flip<<<cdiv(NE, TPB), TPB, 0, stream>>>(B.outb, yb);

  // --- final 1x1 linear over concat + residual + scatter to (b,c,t,f) ---
  k_concat<<<cdiv(M * 128, TPB), TPB, 0, stream>>>(yf, yb, cat_bf, M);
  {
    int waves = (M / 16) * (64 / 16);
    k_gemm_wmma<true><<<cdiv(waves, 4), 128, 0, stream>>>(cat_bf, w_flT, fl_b, B.outb, M, 64, 128);
  }
  k_scatter<<<cdiv(NE, TPB), TPB, 0, stream>>>(B.outb, xf, out, NE);
}
